// MultiHeadAttention_29549374997053
// MI455X (gfx1250) — matrix operand order repaired and checked by reading, not compiled, not run on hardware
//
#include <hip/hip_runtime.h>
#include <hip/hip_bf16.h>
#include <math.h>

typedef _Float16 half_t;
typedef __attribute__((ext_vector_type(16))) _Float16 v16h;
typedef __attribute__((ext_vector_type(8)))  _Float16 v8h;
typedef __attribute__((ext_vector_type(8)))  float    v8f;

#define NHEADS 8
#define HD     64
#define NQ     4096
#define NV     4096
#define DMODEL 512

// ---------------- WMMA fragment helpers (wave32, gfx1250 layouts) ----------------

__device__ __forceinline__ v16h cat8(v8h lo, v8h hi) {
  return __builtin_shufflevector(lo, hi, 0,1,2,3,4,5,6,7,8,9,10,11,12,13,14,15);
}

// A-matrix 16x32 f16, row-major source (lda halves). Lane layout per ISA:
// lanes 0-15: K = kbase+0..7 and kbase+16..23 ; lanes 16-31: K = kbase+8..15 and kbase+24..31
__device__ __forceinline__ v16h load_a_frag(const half_t* A, int lda, int kbase, int l15, int lh) {
  const half_t* p = A + (size_t)l15 * lda + kbase + 8 * lh;
  v8h lo = *(const v8h*)p;
  v8h hi = *(const v8h*)(p + 16);
  return cat8(lo, hi);
}

// B-matrix 32x16 f16 from an N-major source Bt[N][K] (ldb halves).
// lanes 0-15: K = kbase+0..7 and kbase+16..23 ; lanes 16-31: K = kbase+8..15 and kbase+24..31 ; N = lane&15
__device__ __forceinline__ v16h load_b_frag(const half_t* Bt, int ldb, int kbase, int l15, int lh) {
  const half_t* p = Bt + (size_t)l15 * ldb + kbase + 8 * lh;
  v8h lo = *(const v8h*)p;
  v8h hi = *(const v8h*)(p + 16);
  return cat8(lo, hi);
}

__device__ __forceinline__ v8f wmma_f16(v16h a, v16h b, v8f c) {
  return __builtin_amdgcn_wmma_f32_16x16x32_f16(false, a, false, b, (short)0, c, false, false);
}

// ---------------- Prep kernels ----------------

__global__ void f32_to_f16_kernel(const float* __restrict__ src, half_t* __restrict__ dst, int n) {
  int i = blockIdx.x * blockDim.x + threadIdx.x;
  int stride = gridDim.x * blockDim.x;
  for (; i < n; i += stride) dst[i] = (half_t)src[i];
}

// src: [batch][rows][cols] f32  ->  dst: [batch][cols][rows] f16 (batch = blockIdx.y)
__global__ void transpose_f16_kernel(const float* __restrict__ src, half_t* __restrict__ dst,
                                     int rows, int cols) {
  size_t base = (size_t)blockIdx.y * rows * cols;
  int n = rows * cols;
  int i = blockIdx.x * blockDim.x + threadIdx.x;
  int stride = gridDim.x * blockDim.x;
  for (; i < n; i += stride) {
    int r = i / cols, c = i % cols;
    dst[base + (size_t)c * rows + r] = (half_t)src[base + i];
  }
}

// ---------------- Per-head projection GEMM: Out[h] = X(4096x512) @ W[h](512x64) ----------------
// WT: [H][64][512] (N-major). storeT=1 writes Out as [H][64][4096] (for V).
// One wave computes a 32x32 macro-tile (2x2 WMMA tiles).

__global__ __launch_bounds__(256) void proj_kernel(const half_t* __restrict__ X,
                                                   const half_t* __restrict__ WT,
                                                   half_t* __restrict__ Out, int storeT) {
  int lane = threadIdx.x & 31;
  int wave = threadIdx.x >> 5;
  int l15 = lane & 15, lh = lane >> 4;
  int h = blockIdx.y;
  int tile = blockIdx.x * 8 + wave;     // 256 macro-tiles per head
  int mt = tile >> 1;                   // 128 m macro-tiles (32 rows each)
  int nt = tile & 1;                    // 2 n macro-tiles (32 cols each)
  const half_t* A  = X  + (size_t)mt * 32 * DMODEL;
  const half_t* Bt = WT + (size_t)h * HD * DMODEL + (size_t)nt * 32 * DMODEL;
  v8f c[2][2] = {};
  for (int k = 0; k < DMODEL; k += 32) {
    v16h a0 = load_a_frag(A,                 DMODEL, k, l15, lh);
    v16h a1 = load_a_frag(A + 16 * DMODEL,   DMODEL, k, l15, lh);
    v16h b0 = load_b_frag(Bt,                DMODEL, k, l15, lh);
    v16h b1 = load_b_frag(Bt + 16 * DMODEL,  DMODEL, k, l15, lh);
    c[0][0] = wmma_f16(a0, b0, c[0][0]);
    c[0][1] = wmma_f16(a0, b1, c[0][1]);
    c[1][0] = wmma_f16(a1, b0, c[1][0]);
    c[1][1] = wmma_f16(a1, b1, c[1][1]);
  }
  for (int u = 0; u < 2; ++u)
    for (int v = 0; v < 2; ++v) {
      if (!storeT) {
        half_t* C = Out + (size_t)h * NQ * HD + (size_t)(mt * 32 + u * 16) * HD + nt * 32 + v * 16;
        for (int r = 0; r < 8; ++r) {
          int m = r + 8 * lh;
          C[(size_t)m * HD + l15] = (half_t)c[u][v][r];
        }
      } else {
        half_t* C = Out + (size_t)h * HD * NQ + (size_t)(nt * 32 + v * 16) * NQ + mt * 32 + u * 16;
        for (int r = 0; r < 8; ++r) {
          int m = r + 8 * lh;
          C[(size_t)l15 * NQ + m] = (half_t)c[u][v][r];
        }
      }
    }
}

// ---------------- Flash attention: one wave per (head, 32-query group) ----------------

__global__ __launch_bounds__(256) void attn_kernel(const half_t* __restrict__ Qh,   // [H][NQ][HD]
                                                   const half_t* __restrict__ Kh,   // [H][NV][HD]
                                                   const half_t* __restrict__ VhT,  // [H][HD][NV]
                                                   half_t* __restrict__ Concat) {   // [NQ][DMODEL]
  __shared__ __align__(16) half_t pLds[8][32][32];   // 16 KB: per-wave 32x32 P staging
  int lane = threadIdx.x & 31;
  int wave = threadIdx.x >> 5;
  int l15 = lane & 15, lh = lane >> 4;
  int h = blockIdx.y;
  int qg = blockIdx.x * 8 + wave;       // 0..127, 32 queries each

  const half_t* Q  = Qh  + (size_t)h * NQ * HD + (size_t)qg * 32 * HD;
  const half_t* K  = Kh  + (size_t)h * NV * HD;
  const half_t* Vt = VhT + (size_t)h * HD * NV;

  v16h aq[2][2];
  for (int u = 0; u < 2; ++u) {
    aq[u][0] = load_a_frag(Q + (size_t)u * 16 * HD, HD, 0,  l15, lh);
    aq[u][1] = load_a_frag(Q + (size_t)u * 16 * HD, HD, 32, l15, lh);
  }

  float mrow[2][8], lrow[2][8];
  v8f acc[2][4] = {};
  for (int u = 0; u < 2; ++u)
    for (int r = 0; r < 8; ++r) { mrow[u][r] = -__builtin_inff(); lrow[u][r] = 0.f; }

  const float scale = 0.125f;  // 1/sqrt(64)

  for (int j0 = 0; j0 < NV; j0 += 32) {
    // K fragments for key tiles [j0, j0+16) and [j0+16, j0+32), shared by both query tiles
    v16h bk[2][2];
    bk[0][0] = load_b_frag(K + (size_t)j0 * HD,        HD, 0,  l15, lh);
    bk[0][1] = load_b_frag(K + (size_t)j0 * HD,        HD, 32, l15, lh);
    bk[1][0] = load_b_frag(K + (size_t)(j0 + 16) * HD, HD, 0,  l15, lh);
    bk[1][1] = load_b_frag(K + (size_t)(j0 + 16) * HD, HD, 32, l15, lh);

    v8f s[2][2] = {};
    for (int u = 0; u < 2; ++u)
      for (int t2 = 0; t2 < 2; ++t2) {
        s[u][t2] = wmma_f16(aq[u][0], bk[t2][0], s[u][t2]);
        s[u][t2] = wmma_f16(aq[u][1], bk[t2][1], s[u][t2]);
      }

    // online softmax per query tile (row m = r + 8*lh lives in one 16-lane half)
    for (int u = 0; u < 2; ++u) {
      float p0[8], p1[8], alpha[8];
      for (int r = 0; r < 8; ++r) { p0[r] = s[u][0][r] * scale; p1[r] = s[u][1][r] * scale; }
      for (int r = 0; r < 8; ++r) {
        float tm = fmaxf(p0[r], p1[r]);
        for (int off = 1; off < 16; off <<= 1)
          tm = fmaxf(tm, __shfl_xor(tm, off, 16));
        float mnew = fmaxf(mrow[u][r], tm);
        alpha[r] = __expf(mrow[u][r] - mnew);
        mrow[u][r] = mnew;
        p0[r] = __expf(p0[r] - mnew);
        p1[r] = __expf(p1[r] - mnew);
        float rs = p0[r] + p1[r];
        for (int off = 1; off < 16; off <<= 1)
          rs += __shfl_xor(rs, off, 16);
        lrow[u][r] = lrow[u][r] * alpha[r] + rs;
      }
      for (int t = 0; t < 4; ++t)
        for (int r = 0; r < 8; ++r)
          acc[u][t][r] *= alpha[r];
      // relayout P: C-fragment -> LDS rows [u*16 .. u*16+15]
      for (int r = 0; r < 8; ++r) {
        int m = r + 8 * lh;
        pLds[wave][u * 16 + m][l15]      = (half_t)p0[r];
        pLds[wave][u * 16 + m][16 + l15] = (half_t)p1[r];
      }
    }

    v16h ap0 = load_a_frag(&pLds[wave][0][0],  32, 0, l15, lh);  // DS in-order within wave
    v16h ap1 = load_a_frag(&pLds[wave][16][0], 32, 0, l15, lh);

    for (int t = 0; t < 4; ++t) {
      v16h bv = load_b_frag(Vt + (size_t)(t * 16) * NV, NV, j0, l15, lh);
      acc[0][t] = wmma_f16(ap0, bv, acc[0][t]);
      acc[1][t] = wmma_f16(ap1, bv, acc[1][t]);
    }
  }

  for (int u = 0; u < 2; ++u)
    for (int r = 0; r < 8; ++r) {
      int m = r + 8 * lh;
      float inv = 1.f / lrow[u][r];
      half_t* Crow = Concat + (size_t)(qg * 32 + u * 16 + m) * DMODEL + h * HD;
      for (int t = 0; t < 4; ++t)
        Crow[t * 16 + l15] = (half_t)(acc[u][t][r] * inv);
    }
}

// ---------------- Output GEMM: out(4096x512) = Concat @ Wout, f32 store ----------------
// One wave computes a 32x32 macro-tile.

__global__ __launch_bounds__(256) void out_gemm_kernel(const half_t* __restrict__ A,   // [NQ][DMODEL]
                                                       const half_t* __restrict__ Bt,  // [DOUT][DMODEL]
                                                       float* __restrict__ C) {
  int lane = threadIdx.x & 31;
  int wave = threadIdx.x >> 5;
  int l15 = lane & 15, lh = lane >> 4;
  int tile = blockIdx.x * 8 + wave;     // 2048 macro-tiles
  int mt = tile >> 4;                   // 128 m macro-tiles
  int nt = tile & 15;                   // 16 n macro-tiles
  const half_t* Ar = A  + (size_t)mt * 32 * DMODEL;
  const half_t* Br = Bt + (size_t)nt * 32 * DMODEL;
  v8f c[2][2] = {};
  for (int k = 0; k < DMODEL; k += 32) {
    v16h a0 = load_a_frag(Ar,                DMODEL, k, l15, lh);
    v16h a1 = load_a_frag(Ar + 16 * DMODEL,  DMODEL, k, l15, lh);
    v16h b0 = load_b_frag(Br,                DMODEL, k, l15, lh);
    v16h b1 = load_b_frag(Br + 16 * DMODEL,  DMODEL, k, l15, lh);
    c[0][0] = wmma_f16(a0, b0, c[0][0]);
    c[0][1] = wmma_f16(a0, b1, c[0][1]);
    c[1][0] = wmma_f16(a1, b0, c[1][0]);
    c[1][1] = wmma_f16(a1, b1, c[1][1]);
  }
  for (int u = 0; u < 2; ++u)
    for (int v = 0; v < 2; ++v) {
      float* Cp = C + (size_t)(mt * 32 + u * 16) * DMODEL + nt * 32 + v * 16;
      for (int r = 0; r < 8; ++r) {
        int m = r + 8 * lh;
        Cp[(size_t)m * DMODEL + l15] = c[u][v][r];
      }
    }
}

// ---------------- Host launch ----------------

extern "C" void kernel_launch(void* const* d_in, const int* in_sizes, int n_in,
                              void* d_out, int out_size, void* d_ws, size_t ws_size,
                              hipStream_t stream) {
  const float* qin  = (const float*)d_in[0];
  const float* kin  = (const float*)d_in[1];
  const float* vin  = (const float*)d_in[2];
  const float* Wqs  = (const float*)d_in[3];
  const float* Wks  = (const float*)d_in[4];
  const float* Wvs  = (const float*)d_in[5];
  const float* Wout = (const float*)d_in[6];
  float* out = (float*)d_out;

  char* p = (char*)d_ws;
  auto alloc = [&](size_t bytes) {
    char* r = p;
    p += (bytes + 255) & ~(size_t)255;
    return r;
  };
  half_t* Xq   = (half_t*)alloc((size_t)NQ * DMODEL * 2);
  half_t* Xk   = (half_t*)alloc((size_t)NV * DMODEL * 2);
  half_t* Xv   = (half_t*)alloc((size_t)NV * DMODEL * 2);
  half_t* WqT  = (half_t*)alloc((size_t)NHEADS * HD * DMODEL * 2);
  half_t* WkT  = (half_t*)alloc((size_t)NHEADS * HD * DMODEL * 2);
  half_t* WvT  = (half_t*)alloc((size_t)NHEADS * HD * DMODEL * 2);
  half_t* WoT  = (half_t*)alloc((size_t)DMODEL * DMODEL * 2);
  half_t* Qh   = (half_t*)alloc((size_t)NHEADS * NQ * HD * 2);
  half_t* Kh   = (half_t*)alloc((size_t)NHEADS * NV * HD * 2);
  half_t* VhT  = (half_t*)alloc((size_t)NHEADS * HD * NV * 2);
  half_t* Conc = (half_t*)alloc((size_t)NQ * DMODEL * 2);

  f32_to_f16_kernel<<<256, 256, 0, stream>>>(qin, Xq, NQ * DMODEL);
  f32_to_f16_kernel<<<256, 256, 0, stream>>>(kin, Xk, NV * DMODEL);
  f32_to_f16_kernel<<<256, 256, 0, stream>>>(vin, Xv, NV * DMODEL);

  transpose_f16_kernel<<<dim3(128, NHEADS), 256, 0, stream>>>(Wqs, WqT, DMODEL, HD);
  transpose_f16_kernel<<<dim3(128, NHEADS), 256, 0, stream>>>(Wks, WkT, DMODEL, HD);
  transpose_f16_kernel<<<dim3(128, NHEADS), 256, 0, stream>>>(Wvs, WvT, DMODEL, HD);
  transpose_f16_kernel<<<dim3(1024, 1), 256, 0, stream>>>(Wout, WoT, DMODEL, DMODEL);

  proj_kernel<<<dim3(32, NHEADS), 256, 0, stream>>>(Xq, WqT, Qh, 0);
  proj_kernel<<<dim3(32, NHEADS), 256, 0, stream>>>(Xk, WkT, Kh, 0);
  proj_kernel<<<dim3(32, NHEADS), 256, 0, stream>>>(Xv, WvT, VhT, 1);

  attn_kernel<<<dim3(16, NHEADS), 256, 0, stream>>>(Qh, Kh, VhT, Conc);

  out_gemm_kernel<<<256, 256, 0, stream>>>(Conc, WoT, out);
}
